// CPLinear_3547642987329
// MI455X (gfx1250) — matrix operand order repaired and checked by reading, not compiled, not run on hardware
//
#include <hip/hip_runtime.h>
#include <hip/hip_fp16.h>

// ---------------------------------------------------------------------------
// CPLinear fused kernel for gfx1250 (MI455X).
//   GEMM: x(8192x2048) @ W(2048x2016) via v_wmma_f32_16x16x32_f16
//   Epilogue: RoPE + rank contraction -> q,k,v
// Roofline: ~272 MB HBM traffic (11.7us floor), 67.6 GFLOP -> compute-bound
// at the 16-bit matrix rate. 16 waves/WG (4 per SIMD32) so L2 load latency
// on the streamed weights is hidden behind other waves' WMMA issue; inner
// loop keeps dual WMMA chains sharing one A fragment with one-K-step-ahead
// B-fragment prefetch.
// ---------------------------------------------------------------------------

typedef _Float16 f16;
typedef __attribute__((ext_vector_type(4)))  f16    v4h;
typedef __attribute__((ext_vector_type(8)))  f16    v8h;
typedef __attribute__((ext_vector_type(16))) f16    v16h;
typedef __attribute__((ext_vector_type(8)))  float  v8f;

#define H_DIM   2048
#define NTOT    2016      // 192 + 16 + 16 + 1536 + 128 + 128
#define NTILES  126       // NTOT / 16
#define KBLKS   64        // H_DIM / 32
#define ROWS    16        // tokens per workgroup (WMMA M)
#define S_LEN   4096
#define NH      16
#define HD      128
#define RQ      12
#define THREADS 512       // 16 wave32s -> 4 waves per SIMD
#define NWAVES  (THREADS / 32)

// packed-weight column map (N in [0,2016)):
//   [0,192)      A_q   (ld 192)
//   [192,208)    A_k   (ld 16)
//   [208,224)    A_v   (ld 16)
//   [224,1760)   B_q   (ld 1536)
//   [1760,1888)  B_k   (ld 128)
//   [1888,2016)  B_v   (ld 128)

// ---------------------------------------------------------------------------
// Prep: pack fp32 weights -> f16 in the WMMA B-operand layout.
// One thread per (ntile, kblock, lane): 16 strided gathers, one 32B store.
// Layout: halfs[ ((nt*KBLKS + kb)*32 + lane) * 16 + e ]
//   lane<16 : N = nt*16+lane,    K = kb*32 + e        (e = 0..15)
//   lane>=16: N = nt*16+lane-16, K = kb*32 + 16 + e
// ---------------------------------------------------------------------------
__global__ __launch_bounds__(256)
void cpl_pack_weights(const float* __restrict__ WAq, const float* __restrict__ WAk,
                      const float* __restrict__ WAv, const float* __restrict__ WBq,
                      const float* __restrict__ WBk, const float* __restrict__ WBv,
                      f16* __restrict__ wp) {
    int gid = blockIdx.x * blockDim.x + threadIdx.x;      // 0 .. 126*64*32-1
    if (gid >= NTILES * KBLKS * 32) return;
    int lane = gid & 31;
    int kb   = (gid >> 5) & (KBLKS - 1);
    int nt   = gid / (32 * KBLKS);
    int n     = nt * 16 + (lane & 15);
    int kbase = kb * 32 + ((lane < 16) ? 0 : 16);

    const float* src; int col, ld;
    if      (n < 192)  { src = WAq; col = n;        ld = 192;  }
    else if (n < 208)  { src = WAk; col = n - 192;  ld = 16;   }
    else if (n < 224)  { src = WAv; col = n - 208;  ld = 16;   }
    else if (n < 1760) { src = WBq; col = n - 224;  ld = 1536; }
    else if (n < 1888) { src = WBk; col = n - 1760; ld = 128;  }
    else               { src = WBv; col = n - 1888; ld = 128;  }

    v16h o;
    #pragma unroll
    for (int e = 0; e < 16; ++e)
        o[e] = (f16)src[(size_t)(kbase + e) * ld + col];
    *(v16h*)(wp + (size_t)gid * 16) = o;
}

// ---------------------------------------------------------------------------
// Main fused kernel: one WG = 16 tokens, 16 waves, full N (2016 cols).
// ---------------------------------------------------------------------------
__global__ __launch_bounds__(THREADS)
void cpl_main(const float* __restrict__ x,
              const f16* __restrict__ wp,
              float* __restrict__ out) {
    extern __shared__ char smem[];
    f16*   xt  = (f16*)smem;                                          // [16][2048] f16, 64KB
    float* res = (float*)(smem + ROWS * H_DIM * 2);                   // [16][2016] f32, 126KB
    float* trg = (float*)(smem + ROWS * H_DIM * 2 + ROWS * NTOT * 4); // [16][64][2], 8KB

    const int tid  = threadIdx.x;
    const int row0 = blockIdx.x * ROWS;           // global token-row base (0..8176)

    // ---- Phase 1a: x tile fp32 -> f16 LDS (16 float4 per thread) ----
    {
        const float4* xf4 = (const float4*)(x + (size_t)row0 * H_DIM);
        for (int i = tid; i < ROWS * H_DIM / 4; i += THREADS) {
            float4 v = xf4[i];
            v4h h;
            h[0] = (f16)v.x; h[1] = (f16)v.y; h[2] = (f16)v.z; h[3] = (f16)v.w;
            *(v4h*)(xt + i * 4) = h;
        }
    }
    // ---- Phase 1b: RoPE sincos table: pos = row mod 4096, f = 0..63 ----
    for (int i = tid; i < ROWS * 64; i += THREADS) {
        int t = i >> 6, f = i & 63;
        int pos = (row0 + t) & (S_LEN - 1);
        // inv_freq = 10000^(-f/64) = exp2(-f * log2(10000)/64)
        float ang = (float)pos * exp2f(-(float)f * (13.2877123795494f / 64.0f));
        trg[i * 2 + 0] = cosf(ang);
        trg[i * 2 + 1] = sinf(ang);
    }
    __syncthreads();

    // ---- Phase 2: GEMM. 63 N-tile pairs; wave w handles pairs w, w+16, ...
    //      Two WMMA chains share one A fragment; B fragments are prefetched
    //      one K-step ahead into separate live values (software pipeline). ----
    {
        const int wave = tid >> 5;
        const int lane = tid & 31;
        const int arow = lane & 15;                      // A-matrix M row
        const int koff = (lane < 16) ? 0 : 16;           // contiguous K lane split (matches packed B)
        for (int p = wave; p < NTILES / 2; p += NWAVES) {
            const int nt0 = 2 * p;
            v8f acc0 = {};
            v8f acc1 = {};
            const f16* bp = wp + ((size_t)nt0 * KBLKS * 32 + lane) * 16;
            // prologue: fetch kb=0 fragments for both tiles
            v16h b0 = *(const v16h*)(bp);
            v16h b1 = *(const v16h*)(bp + (size_t)KBLKS * 32 * 16);
            #pragma unroll 2
            for (int kb = 0; kb < KBLKS; ++kb) {
                // prefetch next K-step (branchless wrap; last iter re-reads
                // kb=0 from L2 -- harmless, keeps the body straight-line)
                const int kn = (kb + 1) & (KBLKS - 1);
                v16h nb0 = *(const v16h*)(bp + (size_t)kn * 32 * 16);
                v16h nb1 = *(const v16h*)(bp + (size_t)kn * 32 * 16
                                             + (size_t)KBLKS * 32 * 16);
                // A fragment: K{k0..k0+15} of row arow
                int k0 = kb * 32 + koff;
                v8h a0 = *(const v8h*)(xt + arow * H_DIM + k0);
                v8h a1 = *(const v8h*)(xt + arow * H_DIM + k0 + 8);
                v16h a = __builtin_shufflevector(a0, a1,
                    0, 1, 2, 3, 4, 5, 6, 7, 8, 9, 10, 11, 12, 13, 14, 15);
                acc0 = __builtin_amdgcn_wmma_f32_16x16x32_f16(
                    false, a, false, b0, (short)0, acc0, false, false);
                acc1 = __builtin_amdgcn_wmma_f32_16x16x32_f16(
                    false, a, false, b1, (short)0, acc1, false, false);
                b0 = nb0;
                b1 = nb1;
            }
            // C/D layout: VGPR j -> row j (lanes 0-15) or row j+8 (lanes 16-31)
            int col   = nt0 * 16 + (lane & 15);
            int rbase = (lane < 16) ? 0 : 8;
            #pragma unroll
            for (int j = 0; j < 8; ++j) {
                res[(rbase + j) * NTOT + col]      = acc0[j];
                res[(rbase + j) * NTOT + col + 16] = acc1[j];
            }
        }
    }
    __syncthreads();

    // ---- Phase 3: RoPE + rank contraction epilogue, coalesced q/k/v stores ----
    const size_t KOFF = (size_t)2 * S_LEN * NH * HD;     // 16,777,216
    const size_t VOFF = 2 * KOFF;
    for (int idx = tid; idx < ROWS * NH * HD; idx += THREADS) {
        int d = idx & (HD - 1);
        int h = (idx >> 7) & (NH - 1);
        int t = idx >> 11;
        const float* r = res + t * NTOT;
        float cs = trg[(t * 64 + (d & 63)) * 2 + 0];
        float sn = trg[(t * 64 + (d & 63)) * 2 + 1];
        int   dp  = (d < 64) ? d + 64 : d - 64;
        float sgn = (d < 64) ? -1.0f : 1.0f;

        // q[t,h,d] = (1/12) * sum_r A_q[t,h,r] * rope(B_q)[t,r,d]
        float acc = 0.0f;
        #pragma unroll
        for (int rr = 0; rr < RQ; ++rr) {
            float b1 = r[224 + rr * HD + d];
            float b2 = r[224 + rr * HD + dp];
            acc = fmaf(r[h * RQ + rr], fmaf(b1, cs, sgn * b2 * sn), acc);
        }
        size_t o = ((size_t)(row0 + t) * NH + h) * HD + d;
        out[o] = acc * (1.0f / 12.0f);

        // k[t,h,d] = A_k[t,h] * rope(B_k)[t,d]
        {
            float b1 = r[1760 + d], b2 = r[1760 + dp];
            out[KOFF + o] = r[192 + h] * fmaf(b1, cs, sgn * b2 * sn);
        }
        // v[t,h,d] = A_v[t,h] * B_v[t,d]
        out[VOFF + o] = r[208 + h] * r[1888 + d];
    }
}

// ---------------------------------------------------------------------------
extern "C" void kernel_launch(void* const* d_in, const int* in_sizes, int n_in,
                              void* d_out, int out_size, void* d_ws, size_t ws_size,
                              hipStream_t stream) {
    (void)in_sizes; (void)n_in; (void)out_size; (void)ws_size;
    const float* x   = (const float*)d_in[0];
    const float* WAq = (const float*)d_in[1];
    const float* WBq = (const float*)d_in[2];
    const float* WAk = (const float*)d_in[3];
    const float* WBk = (const float*)d_in[4];
    const float* WAv = (const float*)d_in[5];
    const float* WBv = (const float*)d_in[6];
    f16* wp = (f16*)d_ws;                              // 8.25 MB f16 packed weights

    // raise dynamic-LDS cap (gfx1250 WGP has 320KB); harmless if already OK
    (void)hipFuncSetAttribute((const void*)cpl_main,
                              hipFuncAttributeMaxDynamicSharedMemorySize,
                              ROWS * H_DIM * 2 + ROWS * NTOT * 4 + ROWS * 64 * 2 * 4);

    // 1) pack weights: 126*64*32 = 258048 threads
    cpl_pack_weights<<<(NTILES * KBLKS * 32) / 256, 256, 0, stream>>>(
        WAq, WAk, WAv, WBq, WBk, WBv, wp);

    // 2) fused GEMM + RoPE + contraction: 8192/16 = 512 workgroups
    size_t lds = ROWS * H_DIM * 2 + ROWS * NTOT * 4 + ROWS * 64 * 2 * 4; // 202752 B
    cpl_main<<<512, THREADS, lds, stream>>>(x, wp, (float*)d_out);
}
